// MultiHeadAttention_21723944583820
// MI455X (gfx1250) — matrix operand order repaired and checked by reading, not compiled, not run on hardware
//
#include <hip/hip_runtime.h>

// ---------------- problem constants (fixed by reference) ----------------
constexpr int BB = 4;        // batch
constexpr int SS = 2048;     // sequence
constexpr int DD = 1024;     // hidden
constexpr int HH = 16;       // heads
constexpr int HD = 64;       // head dim
constexpr int MM = BB * SS;  // 8192 tokens

typedef __attribute__((ext_vector_type(16))) __bf16 v16bf;
typedef __attribute__((ext_vector_type(8)))  __bf16 v8bf;
typedef __attribute__((ext_vector_type(8)))  float  v8f;
typedef __attribute__((ext_vector_type(4)))  unsigned int u32x4;
typedef __attribute__((ext_vector_type(8)))  int i32x8;
typedef __attribute__((ext_vector_type(4)))  int i32x4;

// ---- TDM availability probe (this toolchain: 6-arg builtin, clang-23 lane) ----
#ifndef HAVE_TDM
#if defined(__has_builtin)
#if __has_builtin(__builtin_amdgcn_tensor_load_to_lds) && __has_builtin(__builtin_amdgcn_s_wait_tensorcnt)
#define HAVE_TDM 1
#else
#define HAVE_TDM 0
#endif
#else
#define HAVE_TDM 0
#endif
#endif

__device__ __forceinline__ v16bf cat8(v8bf lo, v8bf hi) {
    return __builtin_shufflevector(lo, hi, 0,1,2,3,4,5,6,7,8,9,10,11,12,13,14,15);
}

__device__ __forceinline__ v8f wmma_bf16(v16bf a, v16bf b, v8f c) {
    return __builtin_amdgcn_wmma_f32_16x16x32_bf16(
        false, a, false, b, (short)0, c, false, false);
}

// wave32 half-swap (lane <-> lane^16) as pure VALU: v_permlanex16 identity selects
__device__ __forceinline__ float xhalf(float v) {
    int i = __builtin_bit_cast(int, v);
    int r = __builtin_amdgcn_permlanex16(i, i, 0x76543210, 0xfedcba98, false, false);
    return __builtin_bit_cast(float, r);
}

__device__ __forceinline__ unsigned xhalf_u(unsigned v) {
    return (unsigned)__builtin_amdgcn_permlanex16((int)v, (int)v, 0x76543210, 0xfedcba98,
                                                  false, false);
}

// raw v_exp_f32: softmax args are <= 0 and subnormal outputs are negligible in the sum,
// so skip libm's denorm-guard sequence (saves ~5 VALU per exp on the critical path)
__device__ __forceinline__ float fast_exp2(float x) {
#if defined(__has_builtin) && __has_builtin(__builtin_amdgcn_exp2f)
    return __builtin_amdgcn_exp2f(x);
#else
    return exp2f(x);
#endif
}

// pack two floats into a bf16x2 word (RNE via scalar __bf16 converts)
__device__ __forceinline__ unsigned pack_bf16(float lo, float hi) {
    unsigned a = (unsigned)__builtin_bit_cast(unsigned short, (__bf16)lo);
    unsigned b = (unsigned)__builtin_bit_cast(unsigned short, (__bf16)hi);
    return a | (b << 16);
}

// ---- Tensor Data Mover: 2D tile (rows x contiguous run) global -> LDS ----
// D# per CDNA5 ISA ch.8: group0 {count=1, lds_addr, global_addr, type=2},
// group1 {data_size=2B, tensor_dim0/1, tile_dim0/1, tensor_dim0_stride}.
__device__ __forceinline__ void tdm_load_2d(unsigned lds_off, const void* gptr,
                                            unsigned tile_d0, unsigned tile_d1,
                                            unsigned long long stride0_elems) {
#if HAVE_TDM
    unsigned long long ga = (unsigned long long)gptr;
    u32x4 g0;
    g0[0] = 1u;                                        // count=1, user mode
    g0[1] = lds_off;                                   // LDS byte address
    g0[2] = (unsigned)ga;                              // global_addr[31:0]
    g0[3] = (unsigned)((ga >> 32) & 0x01FFFFFFull) | (2u << 30);  // addr[56:32] | type=2
    i32x8 g1;
    g1[0] = (int)(1u << 16);                           // workgroup_mask=0, data_size=1 (2B)
    g1[1] = (int)((tile_d0 & 0xFFFFu) << 16);          // tensor_dim0[15:0] (== tile)
    g1[2] = (int)((tile_d0 >> 16) | ((tile_d1 & 0xFFFFu) << 16));  // dim0 hi | dim1 lo
    g1[3] = (int)((tile_d1 >> 16) | (tile_d0 << 16));  // dim1 hi | tile_dim0
    g1[4] = (int)(tile_d1 & 0xFFFFu);                  // tile_dim1 (tile_dim2=0)
    g1[5] = (int)(unsigned)stride0_elems;              // tensor_dim0_stride lo32
    g1[6] = (int)((unsigned)(stride0_elems >> 32) & 0xFFFFu);
    g1[7] = 0;
    i32x4 z4; z4[0] = z4[1] = z4[2] = z4[3] = 0;       // D# groups 2/3 unused (2D tile)
    i32x8 z8;
#pragma unroll
    for (int i = 0; i < 8; ++i) z8[i] = 0;
    __builtin_amdgcn_tensor_load_to_lds(g0, g1, z4, z4, z8, 0);
#else
    (void)lds_off; (void)gptr; (void)tile_d0; (void)tile_d1; (void)stride0_elems;
#endif
}

__device__ __forceinline__ void tdm_wait0() {
#if HAVE_TDM
    __builtin_amdgcn_s_wait_tensorcnt((short)0);
#endif
}

__device__ __forceinline__ unsigned lds_addr_of(const void* p) {
    return (unsigned)(unsigned long long)p;   // LDS aperture: low 32 bits = LDS byte offset
}

// ---------------- fp32 -> bf16 conversion ----------------
__global__ void cvt_bf16(const float* __restrict__ in, __bf16* __restrict__ out, int n) {
    int i = blockIdx.x * blockDim.x + threadIdx.x;
    if (i < n) out[i] = (__bf16)in[i];
}

// ---------------- GEMM: C[M,N] = A[M,K](bf16) @ W[N,K]^T(bf16) + bias ----------------
// Block = 8 waves; wave computes 32x64 (2 m-tiles x 4 n-tiles, 8 WMMA / k-step).
// Block covers 256 rows x 64 cols; the shared W slab (64 rows x 32 k) is staged in LDS
// by the TDM (double buffered) so each k-step costs one DMA instead of 8x redundant loads.
// MODE 0: out bf16 [B,H,S,hd] (Q,K) | MODE 1: out bf16 [B,H,hd,S] (V^T) | MODE 2: fp32 [M,N]
template <int MODE>
__global__ void gemm_bt(const __bf16* __restrict__ A, const __bf16* __restrict__ W,
                        const float* __restrict__ bias, void* __restrict__ out,
                        int M, int N, int K, float scale) {
    constexpr int NBUF = HAVE_TDM ? 2 : 1;
    __shared__ __align__(16) __bf16 sW[NBUF][64 * 32];

    const int lane = threadIdx.x & 31;
    const int half = lane >> 4;
    const int lm   = lane & 15;
    const int wid  = threadIdx.x >> 5;
    const int tid  = threadIdx.x;
    const int nstrips = N >> 6;
    const int mblk = blockIdx.x / nstrips;
    const int ns   = blockIdx.x % nstrips;
    const int m0 = mblk * 256 + wid * 32;
    const int n0 = ns << 6;

    v8f acc[2][4];
#pragma unroll
    for (int u = 0; u < 2; ++u)
#pragma unroll
        for (int t = 0; t < 4; ++t)
#pragma unroll
            for (int r = 0; r < 8; ++r) acc[u][t][r] = 0.f;

    const __bf16* ap0 = A + (size_t)(m0 + lm) * K;        // m-tile 0 row
    const __bf16* ap1 = ap0 + (size_t)16 * K;             // m-tile 1 row
    const __bf16* wbase = W + (size_t)n0 * K;

#if HAVE_TDM
    if (wid == 0) tdm_load_2d(lds_addr_of(&sW[0][0]), wbase, 32, 64, (unsigned)K);
#endif

    for (int k0 = 0, c = 0; k0 < K; k0 += 32, ++c) {
        const int buf = c & (NBUF - 1);
#if HAVE_TDM
        if (wid == 0) tdm_wait0();
        __syncthreads();                        // slab c visible; slab c+1's buffer free
        if (wid == 0 && k0 + 32 < K)
            tdm_load_2d(lds_addr_of(&sW[buf ^ 1][0]), wbase + (k0 + 32), 32, 64, (unsigned)K);
#else
        __syncthreads();                        // previous reads done
        {   // cooperative copy: 256 threads x 16B = 4KB slab
            int row = tid >> 2, col = (tid & 3) * 8;
            *(v8bf*)&sW[0][row * 32 + col] = *(const v8bf*)(wbase + (size_t)row * K + k0 + col);
        }
        __syncthreads();
#endif
        // A fragments (global; per-lane k pattern (e/8)*16 + half*8 + e%8)
        v16bf a0 = cat8(*(const v8bf*)(ap0 + k0 + half * 8),
                        *(const v8bf*)(ap0 + k0 + 16 + half * 8));
        v16bf a1 = cat8(*(const v8bf*)(ap1 + k0 + half * 8),
                        *(const v8bf*)(ap1 + k0 + 16 + half * 8));
#pragma unroll
        for (int t = 0; t < 4; ++t) {
            // B fragment from LDS: W row (n0+t*16+lm), k pattern (e/8)*16 + half*8 + e%8
            v16bf b = cat8(*(const v8bf*)&sW[buf][(t * 16 + lm) * 32 + half * 8],
                           *(const v8bf*)&sW[buf][(t * 16 + lm) * 32 + 16 + half * 8]);
            acc[0][t] = wmma_bf16(a0, b, acc[0][t]);
            acc[1][t] = wmma_bf16(a1, b, acc[1][t]);
        }
    }

#pragma unroll
    for (int u = 0; u < 2; ++u) {
        const int mb = m0 + u * 16;
#pragma unroll
        for (int t = 0; t < 4; ++t) {
            const int n_g = n0 + t * 16 + lm;
            const float bn = bias[n_g];
            if constexpr (MODE == 2) {
                float* o = (float*)out;
#pragma unroll
                for (int r = 0; r < 8; ++r) {
                    int m_g = mb + half * 8 + r;          // D layout: m = r + 8*half
                    o[(size_t)m_g * N + n_g] = (acc[u][t][r] + bn) * scale;
                }
            } else if constexpr (MODE == 0) {
                __bf16* o = (__bf16*)out;
                const int h = n_g / HD, d = n_g % HD;
#pragma unroll
                for (int r = 0; r < 8; ++r) {
                    int m_g = mb + half * 8 + r;
                    int b_i = m_g / SS, s = m_g % SS;
                    size_t idx = (((size_t)(b_i * HH + h)) * SS + s) * HD + d;
                    o[idx] = (__bf16)((acc[u][t][r] + bn) * scale);
                }
            } else {  // MODE 1: V^T [B,H,hd,S]; 8 rows contiguous in s -> one 16B store
                __bf16* o = (__bf16*)out;
                const int h = n_g / HD, d = n_g % HD;
                const int m_b = mb + half * 8;
                const int b_i = m_b / SS, s0 = m_b % SS;
                size_t idx0 = (((size_t)(b_i * HH + h)) * HD + d) * SS + s0;
                v8bf pk;
#pragma unroll
                for (int r = 0; r < 8; ++r) pk[r] = (__bf16)((acc[u][t][r] + bn) * scale);
                *(v8bf*)(o + idx0) = pk;
            }
        }
    }
}

// ---------------- flash attention ----------------
// Block = 8 waves = 128 queries of ONE head -> all waves share the K/V stream.
// 32-key chunks of K (32x64) and V^T (64x32) are TDM-staged into LDS (double buffered).
// Computes S^T = K_chunk x Q^T so per-query softmax stats are lane-local (q = lane%16).
// Q pre-scaled by (1/sqrt(hd))*log2(e); exp2 is the softmax exponential.
__global__ void attn_kernel(const __bf16* __restrict__ Q, const __bf16* __restrict__ Kmat,
                            const __bf16* __restrict__ Vt, __bf16* __restrict__ ctx) {
    constexpr int NBUF = HAVE_TDM ? 2 : 1;
    __shared__ __align__(16) __bf16 sK[NBUF][32 * 64];   // [key][d]
    __shared__ __align__(16) __bf16 sV[NBUF][64 * 32];   // [d][key]

    const int lane = threadIdx.x & 31;
    const int half = lane >> 4;
    const int lm   = lane & 15;
    const int wid  = threadIdx.x >> 5;
    const int tid  = threadIdx.x;

    const int blocks_per_head = SS / 128;                // 16
    const int bh = blockIdx.x / blocks_per_head;
    const int q0 = (blockIdx.x % blocks_per_head) * 128 + wid * 16;
    const int b_i = bh / HH;
    const int h   = bh % HH;

    const __bf16* qbase = Q    + (size_t)bh * SS * HD;
    const __bf16* kbase = Kmat + (size_t)bh * SS * HD;
    const __bf16* vbase = Vt   + (size_t)bh * HD * SS;

    // Q^T B-fragments (k = d = (e/8)*16 + half*8 + e%8): two 16B loads each, kept in regs.
    const __bf16* qrow = qbase + (size_t)(q0 + lm) * HD;
    const v16bf qb0 = cat8(*(const v8bf*)(qrow + half * 8), *(const v8bf*)(qrow + 16 + half * 8));
    const v16bf qb1 = cat8(*(const v8bf*)(qrow + 32 + half * 8), *(const v8bf*)(qrow + 48 + half * 8));

    v8f o[4];
#pragma unroll
    for (int t = 0; t < 4; ++t)
#pragma unroll
        for (int r = 0; r < 8; ++r) o[t][r] = 0.f;
    float m_q = -1e30f, l_q = 0.f;

#if HAVE_TDM
    if (wid == 0) {
        tdm_load_2d(lds_addr_of(&sK[0][0]), kbase, 64, 32, HD);
        tdm_load_2d(lds_addr_of(&sV[0][0]), vbase, 32, 64, SS);
    }
#endif

    for (int j0 = 0, c = 0; j0 < SS; j0 += 32, ++c) {
        const int buf = c & (NBUF - 1);
#if HAVE_TDM
        if (wid == 0) tdm_wait0();
        __syncthreads();
        if (wid == 0 && j0 + 32 < SS) {
            tdm_load_2d(lds_addr_of(&sK[buf ^ 1][0]), kbase + (size_t)(j0 + 32) * HD, 64, 32, HD);
            tdm_load_2d(lds_addr_of(&sV[buf ^ 1][0]), vbase + (j0 + 32), 32, 64, SS);
        }
#else
        __syncthreads();
        {   // cooperative copies: 4KB each, 16B per thread
            int kr = tid >> 3, kc = (tid & 7) * 8;       // sK: 32 rows x 64
            *(v8bf*)&sK[0][kr * 64 + kc] = *(const v8bf*)(kbase + (size_t)(j0 + kr) * HD + kc);
            int vr = tid >> 2, vc = (tid & 3) * 8;       // sV: 64 rows x 32
            *(v8bf*)&sV[0][vr * 32 + vc] = *(const v8bf*)(vbase + (size_t)vr * SS + j0 + vc);
        }
        __syncthreads();
#endif

        // ---- scores S^T (32 keys x 16 queries): 4 WMMAs, A-frags from LDS ----
        v8f s0 = {}, s1 = {};
        {
            const __bf16* kr0 = &sK[buf][lm * 64];
            v16bf a00 = cat8(*(const v8bf*)(kr0 + half * 8),
                             *(const v8bf*)(kr0 + 16 + half * 8));
            v16bf a01 = cat8(*(const v8bf*)(kr0 + 32 + half * 8),
                             *(const v8bf*)(kr0 + 48 + half * 8));
            s0 = wmma_bf16(a00, qb0, s0);
            s0 = wmma_bf16(a01, qb1, s0);
            const __bf16* kr1 = kr0 + 16 * 64;
            v16bf a10 = cat8(*(const v8bf*)(kr1 + half * 8),
                             *(const v8bf*)(kr1 + 16 + half * 8));
            v16bf a11 = cat8(*(const v8bf*)(kr1 + 32 + half * 8),
                             *(const v8bf*)(kr1 + 48 + half * 8));
            s1 = wmma_bf16(a10, qb0, s1);
            s1 = wmma_bf16(a11, qb1, s1);
        }

        // ---- online softmax (lane-local + one half-swap) ----
        float cmax = -1e30f;
#pragma unroll
        for (int r = 0; r < 8; ++r) cmax = fmaxf(cmax, fmaxf(s0[r], s1[r]));
        cmax = fmaxf(cmax, xhalf(cmax));
        const float newm = fmaxf(m_q, cmax);
        const float alpha = fast_exp2(m_q - newm);
        float p0[8], p1[8];
        float rsum = 0.f;
#pragma unroll
        for (int r = 0; r < 8; ++r) {
            p0[r] = fast_exp2(s0[r] - newm);
            p1[r] = fast_exp2(s1[r] - newm);
            rsum += p0[r] + p1[r];
        }
        rsum += xhalf(rsum);
        l_q = l_q * alpha + rsum;
        m_q = newm;
#pragma unroll
        for (int t = 0; t < 4; ++t)
#pragma unroll
            for (int r = 0; r < 8; ++r) o[t][r] *= alpha;

        // ---- assemble P^T as B-fragment (k = j = half*16 + e, n = q), packed bf16x2 ----
        i32x8 pw;
#pragma unroll
        for (int e2 = 0; e2 < 4; ++e2) {
            unsigned w0 = pack_bf16(p0[2 * e2], p0[2 * e2 + 1]);   // own j-tile0 pair
            unsigned w1 = pack_bf16(p1[2 * e2], p1[2 * e2 + 1]);   // own j-tile1 pair
            unsigned x1 = xhalf_u(w1);                             // other half's tile1
            unsigned x0 = xhalf_u(w0);                             // other half's tile0
            pw[e2]     = (int)(half == 0 ? w0 : x1);               // j = half*16 + 2e2{,+1}
            pw[4 + e2] = (int)(half == 0 ? x0 : w1);               // j = half*16 + 8 + 2e2{,+1}
        }
        v16bf pb = __builtin_bit_cast(v16bf, pw);

        // ---- O^T += V^T x P^T (4 WMMAs, A-frags from LDS) ----
#pragma unroll
        for (int t = 0; t < 4; ++t) {
            const __bf16* vr = &sV[buf][(t * 16 + lm) * 32];
            v16bf av = cat8(*(const v8bf*)(vr + half * 16),
                            *(const v8bf*)(vr + half * 16 + 8));
            o[t] = wmma_bf16(av, pb, o[t]);
        }
    }

    // ---- epilogue: ctx[b, s, h*64 + d] (bf16); 16B packed stores ----
    const float inv_l = 1.f / l_q;
    __bf16* crow = ctx + (size_t)(b_i * SS + q0 + lm) * DD + h * HD;
#pragma unroll
    for (int t = 0; t < 4; ++t) {
        v8bf pk;
#pragma unroll
        for (int r = 0; r < 8; ++r) pk[r] = (__bf16)(o[t][r] * inv_l);
        *(v8bf*)(crow + t * 16 + half * 8) = pk;
    }
}

// ---------------- host-side launcher ----------------
extern "C" void kernel_launch(void* const* d_in, const int* in_sizes, int n_in,
                              void* d_out, int out_size, void* d_ws, size_t ws_size,
                              hipStream_t stream) {
    const float* x  = (const float*)d_in[0];
    const float* Wq = (const float*)d_in[1];
    const float* bq = (const float*)d_in[2];
    const float* Wk = (const float*)d_in[3];
    const float* bk = (const float*)d_in[4];
    const float* Wv = (const float*)d_in[5];
    const float* bv = (const float*)d_in[6];
    const float* Wo = (const float*)d_in[7];
    const float* bo = (const float*)d_in[8];
    float* out = (float*)d_out;

    size_t off = 0;
    auto carve = [&](size_t elems) {
        __bf16* p = (__bf16*)((char*)d_ws + off);
        off += ((elems * sizeof(__bf16)) + 255) & ~(size_t)255;
        return p;
    };
    __bf16* xb   = carve((size_t)MM * DD);
    __bf16* wqb  = carve((size_t)DD * DD);
    __bf16* wkb  = carve((size_t)DD * DD);
    __bf16* wvb  = carve((size_t)DD * DD);
    __bf16* wob  = carve((size_t)DD * DD);
    __bf16* Qb   = carve((size_t)MM * DD);   // [B,H,S,hd]
    __bf16* Kb   = carve((size_t)MM * DD);   // [B,H,S,hd]
    __bf16* Vtb  = carve((size_t)MM * DD);   // [B,H,hd,S]
    __bf16* ctxb = carve((size_t)MM * DD);   // [M,D]
    (void)ws_size; (void)in_sizes; (void)n_in; (void)out_size;

    auto cvt = [&](const float* src, __bf16* dst, size_t n) {
        int blocks = (int)((n + 255) / 256);
        cvt_bf16<<<blocks, 256, 0, stream>>>(src, dst, (int)n);
    };
    cvt(x,  xb,  (size_t)MM * DD);
    cvt(Wq, wqb, (size_t)DD * DD);
    cvt(Wk, wkb, (size_t)DD * DD);
    cvt(Wv, wvb, (size_t)DD * DD);
    cvt(Wo, wob, (size_t)DD * DD);

    const int gemm_blocks = (MM / 256) * (DD / 64);     // 512 blocks x 8 waves
    const float qscale = 0.125f * 1.44269504088896340736f;  // (1/sqrt(hd)) * log2(e)
    gemm_bt<0><<<gemm_blocks, 256, 0, stream>>>(xb, wqb, bq, Qb,  MM, DD, DD, qscale);
    gemm_bt<0><<<gemm_blocks, 256, 0, stream>>>(xb, wkb, bk, Kb,  MM, DD, DD, 1.0f);
    gemm_bt<1><<<gemm_blocks, 256, 0, stream>>>(xb, wvb, bv, Vtb, MM, DD, DD, 1.0f);

    const int attn_blocks = BB * HH * (SS / 128);       // 1024 blocks
    attn_kernel<<<attn_blocks, 256, 0, stream>>>(Qb, Kb, Vtb, ctxb);

    gemm_bt<2><<<gemm_blocks, 256, 0, stream>>>(ctxb, wob, bo, out, MM, DD, DD, 1.0f);
}
